// kspaceMap_14843406975412
// MI455X (gfx1250) — compile-verified, matrix operand order checked by reading, not run on hardware
//
#include <hip/hip_runtime.h>
#include <hip/hip_bf16.h>

// ---------------------------------------------------------------------------
// y[b,o,h,w] = leaky_relu( sum_{h'} g[o,(h-h')&255] * x[b,h',w] )
//   g[o,d]   = (1/256) * sum_c W[o,c] * cos(2*pi*c*d/256)
// out = concat(y, input[:,1,:,:]) -> (8, 257, 256, 256)
// GEMM: M = 65536 (o,h), N = 2048 (b,w), K = 256, bf16 hi/lo split x3 WMMA.
// Prep kernels pre-split operands to bf16 hi/lo in global memory (once, not
// per-block). Hot kernel: B tiles DMA'd straight into padded LDS by the
// Tensor Data Mover (tensor_load_to_lds, TENSORcnt), A gathered from the
// reversed g-row; every WMMA operand VGPR is one aligned ds_load_b32.
// ---------------------------------------------------------------------------

typedef __attribute__((ext_vector_type(16))) __bf16 v16bf;
typedef __attribute__((ext_vector_type(8)))  float  v8f;
typedef __attribute__((ext_vector_type(4)))  unsigned v4u;
typedef __attribute__((ext_vector_type(8)))  int      v8i_;
typedef __attribute__((ext_vector_type(4)))  int      v4i_;

union ABfrag { v16bf v; unsigned short u[16]; unsigned d[8]; };

__device__ __forceinline__ unsigned short f2bf(float f) {
    unsigned u = __float_as_uint(f);
    unsigned r = u + 0x7FFFu + ((u >> 16) & 1u);   // round-to-nearest-even
    return (unsigned short)(r >> 16);
}
__device__ __forceinline__ float bf2f(unsigned short b) {
    return __uint_as_float(((unsigned)b) << 16);
}

// ---- TDM: load 2D tile (32 x 128 of 2B elems, row stride 256 elems) into LDS
// with padding 16B after every 64B row -> 80B padded LDS row stride.
__device__ __forceinline__ void tdm_load_tile(unsigned lds_addr,
                                              const unsigned short* gptr) {
    unsigned long long ga = (unsigned long long)(uintptr_t)gptr;
    v4u g0;
    g0[0] = 1u;                                            // count=1 (valid D#)
    g0[1] = lds_addr;                                      // LDS byte offset
    g0[2] = (unsigned)ga;                                  // global_addr[31:0]
    g0[3] = (unsigned)((ga >> 32) & 0x01FFFFFFu) | (2u << 30);  // [56:32], type=2
    v8i_ g1;
    g1[0] = (int)((1u << 16)      // data_size = 1 -> 2 bytes
                | (1u << 20)      // pad_enable
                | (3u << 22)      // pad_interval: 16 DWORDs (64B) per pad
                | (3u << 25));    // pad_amount: 4 DWORDs (16B)
    g1[1] = (int)(256u << 16);    // tensor_dim0 = 256 (elements)
    g1[2] = (int)(2048u << 16);   // tensor_dim1 = 2048 rows (no OOB in range)
    g1[3] = (int)(32u << 16);     // tile_dim0 = 32 elements (64B)
    g1[4] = 128;                  // tile_dim1 = 128 rows, tile_dim2 = 0
    g1[5] = 256;                  // tensor_dim0_stride = 256 elements
    g1[6] = 0;
    g1[7] = 0;
    v4i_ z4 = {0, 0, 0, 0};       // groups 2/3 unused (2D tile)
    v8i_ z8 = {0, 0, 0, 0, 0, 0, 0, 0};
    __builtin_amdgcn_tensor_load_to_lds(g0, g1, z4, z4, z8, 0);
}

// ---- Prep 1: reversed/duplicated bf16 g-row tables, per channel o.
// Apre[o][0..511]=hiO, [512..1023]=loO, [1024..1535]=hiE, [1536..2047]=loE
//   sAO[j] = g[(255-j)&255],  sAE[j] = g[(-j)&255]
__global__ __launch_bounds__(256) void build_g_kernel(const float* __restrict__ W,
                                                      unsigned short* __restrict__ Apre) {
    const int o = blockIdx.x;
    const int d = threadIdx.x;
    const float PI_OVER_128 = 0.0245436926061702596754894014318f;
    float s = 0.0f;
    for (int c = 0; c < 256; ++c) {
        int m = (c * d) & 255;                       // exact phase reduction
        s += W[o * 256 + c] * cosf((float)m * PI_OVER_128);
    }
    float gv = s * (1.0f / 256.0f);
    unsigned short hi = f2bf(gv);
    unsigned short lo = f2bf(gv - bf2f(hi));
    unsigned short* A = Apre + (size_t)o * 2048;
    int jO = 255 - d;
    A[jO]        = hi;  A[jO + 256]        = hi;     // hiO
    A[512 + jO]  = lo;  A[512 + jO + 256]  = lo;     // loO
    int jE = (d == 0) ? 0 : (256 - d);
    A[1024 + jE] = hi;  A[1024 + jE + 256] = hi;     // hiE
    A[1536 + jE] = lo;  A[1536 + jE + 256] = lo;     // loE
}

// ---- Prep 2: X -> bf16 hi/lo, transposed to [b][w][k] (k contiguous).
__global__ __launch_bounds__(256) void split_x_kernel(const float* __restrict__ in,
                                                      unsigned short* __restrict__ Xhi,
                                                      unsigned short* __restrict__ Xlo) {
    const int bi = blockIdx.x;            // 0..2047
    const int b  = bi >> 8;
    const int w  = bi & 255;
    const int k  = threadIdx.x;
    float gv = in[(size_t)b * 131072 + (size_t)k * 256 + w];
    unsigned short hi = f2bf(gv);
    size_t oidx = (((size_t)b * 256 + w) << 8) + k;  // coalesced writes
    Xhi[oidx] = hi;
    Xlo[oidx] = f2bf(gv - bf2f(hi));
}

// ---- Kernel 3: tiled GEMM. TDM-fed B, reversed-g A, fused leaky-relu.
__global__ __launch_bounds__(256) void kspace_gemm_kernel(
        const unsigned short* __restrict__ Apre,
        const unsigned short* __restrict__ Xhi,
        const unsigned short* __restrict__ Xlo,
        float* __restrict__ out) {
    __shared__ __align__(16) unsigned short sA[4][512];       // hiO,loO,hiE,loE
    __shared__ __align__(16) unsigned short sB[2][2][128][40]; // [buf][mat][n][k]

    const int tid = threadIdx.x;
    const int mt  = blockIdx.x;              // 0..511  (M tiles of 128)
    const int nt  = blockIdx.y;              // 0..15   (N tiles of 128)
    const int o   = mt >> 1;
    const int h0  = (mt & 1) << 7;
    const int b   = nt >> 1;
    const int w0  = (nt & 1) << 7;

    const int wv   = tid >> 5;               // wave 0..7
    const int wv_m = wv >> 1;                // 0..3 -> M sub-pair
    const int wv_n = wv & 1;                 // 0..1 -> N quad
    const int lane = tid & 31;
    const int hi16 = lane >> 4;
    const int mrow = lane & 15;

    const unsigned ldsB = (unsigned)(uintptr_t)&sB[0][0][0][0];
    const unsigned short* xrowH = Xhi + (((size_t)b * 256 + w0) << 8);
    const unsigned short* xrowL = Xlo + (((size_t)b * 256 + w0) << 8);

    // Issue TDM loads for chunk kc into buffer `buf` (one wave only; EXEC is
    // ignored by TDM, but the uniform branch keeps other waves from issuing).
    auto issueB = [&](int buf, int kc) {
        if (wv == 0) {
            const int k0 = kc << 5;
            tdm_load_tile(ldsB + (unsigned)((buf * 2 + 0) * 10240), xrowH + k0);
            tdm_load_tile(ldsB + (unsigned)((buf * 2 + 1) * 10240), xrowL + k0);
        }
    };

    issueB(0, 0);                            // DMA chunk 0 while we stage A

    // Stage per-channel A tables (4 KB) into LDS.
    {
        const unsigned* src = reinterpret_cast<const unsigned*>(Apre + (size_t)o * 2048);
        unsigned* dst = reinterpret_cast<unsigned*>(&sA[0][0]);
#pragma unroll
        for (int i = 0; i < 4; ++i) dst[tid + (i << 8)] = src[tid + (i << 8)];
    }

    // Two A row strips per wave; both share parity (differ by 16).
    const int h_a0 = h0 + (wv_m << 5) + mrow;
    const int par  = h_a0 & 1;
    const unsigned short* pAhi = par ? sA[0] : sA[2];
    const unsigned short* pAlo = par ? sA[1] : sA[3];
    const int joff0 = (par ? 255 : 256) - h_a0;

    v8f acc[2][4];
#pragma unroll
    for (int m = 0; m < 2; ++m)
#pragma unroll
        for (int t = 0; t < 4; ++t) acc[m][t] = v8f{};

    for (int kc = 0; kc < 8; ++kc) {
        if (wv == 0) __builtin_amdgcn_s_wait_tensorcnt(0);   // own TDM done
        __syncthreads();                                     // publish buffer
        if (kc < 7) issueB((kc + 1) & 1, kc + 1);            // overlap next DMA

        const int k0  = kc << 5;
        const int buf = kc & 1;

        // A fragments: VGPR v holds K = kp(v), kp(v)+1; one ds_load_b32 each.
        ABfrag ahi[2], alo[2];
#pragma unroll
        for (int m = 0; m < 2; ++m) {
            const int jb = joff0 - (m << 4) + k0;            // even -> aligned
#pragma unroll
            for (int v = 0; v < 8; ++v) {
                int kp = (v << 1) + ((v >= 4) ? 8 : 0) + (hi16 << 3);
                ahi[m].d[v] = *reinterpret_cast<const unsigned*>(&pAhi[jb + kp]);
                alo[m].d[v] = *reinterpret_cast<const unsigned*>(&pAlo[jb + kp]);
            }
        }

#pragma unroll
        for (int t = 0; t < 4; ++t) {
            const int col = (wv_n << 6) + (t << 4) + mrow;
            ABfrag bhi, blo;
#pragma unroll
            for (int v = 0; v < 8; ++v) {                    // one b32 per VGPR
                int kp = (v << 1) + ((v >= 4) ? 8 : 0) + (hi16 << 3);
                bhi.d[v] = *reinterpret_cast<const unsigned*>(&sB[buf][0][col][kp]);
                blo.d[v] = *reinterpret_cast<const unsigned*>(&sB[buf][1][col][kp]);
            }
#pragma unroll
            for (int m = 0; m < 2; ++m) {
                // hi*hi + hi*lo + lo*hi (drop lo*lo) ~ fp32 accuracy
                acc[m][t] = __builtin_amdgcn_wmma_f32_16x16x32_bf16(
                    false, ahi[m].v, false, bhi.v, (short)0, acc[m][t], false, false);
                acc[m][t] = __builtin_amdgcn_wmma_f32_16x16x32_bf16(
                    false, ahi[m].v, false, blo.v, (short)0, acc[m][t], false, false);
                acc[m][t] = __builtin_amdgcn_wmma_f32_16x16x32_bf16(
                    false, alo[m].v, false, bhi.v, (short)0, acc[m][t], false, false);
            }
        }
    }

    // Epilogue: leaky_relu(0.2) + store. C/D: M = r + 8*hi16, N = lane&15.
    float* __restrict__ obase = out + ((size_t)(b * 257 + o)) * 65536;
#pragma unroll
    for (int m = 0; m < 2; ++m) {
        const int hbase = h0 + (wv_m << 5) + (m << 4) + (hi16 << 3);
#pragma unroll
        for (int t = 0; t < 4; ++t) {
            const int wcol = w0 + (wv_n << 6) + (t << 4) + (lane & 15);
#pragma unroll
            for (int r = 0; r < 8; ++r) {
                float v = acc[m][t][r];
                v = (v >= 0.0f) ? v : 0.2f * v;
                obase[(size_t)(hbase + r) * 256 + wcol] = v;   // lane-coalesced
            }
        }
    }
}

// ---- Kernel 4: pass-through of input[:,1,:,:] into output channel 256
__global__ __launch_bounds__(256) void copy_others_kernel(const float* __restrict__ in,
                                                          float* __restrict__ out) {
    int idx = blockIdx.x * 256 + threadIdx.x;    // float4 index, 131072 total
    int b   = idx >> 14;
    int r   = idx & 16383;
    const float4 v = reinterpret_cast<const float4*>(in + (size_t)b * 131072 + 65536)[r];
    reinterpret_cast<float4*>(out + ((size_t)b * 257 + 256) * 65536)[r] = v;
}

extern "C" void kernel_launch(void* const* d_in, const int* in_sizes, int n_in,
                              void* d_out, int out_size, void* d_ws, size_t ws_size,
                              hipStream_t stream) {
    const float* inp = (const float*)d_in[0];   // (8,2,256,256) fp32
    // d_in[1] = mask -> unused by reference
    const float* W   = (const float*)d_in[2];   // (256,256) fp32
    float* out = (float*)d_out;                 // (8,257,256,256) fp32

    // Workspace layout (3 MB total):
    //   [0, 1MB)   Apre : 256 channels x 2048 ushort (hiO|loO|hiE|loE)
    //   [1MB, 2MB) Xhi  : [b][w][k] bf16
    //   [2MB, 3MB) Xlo  : [b][w][k] bf16
    unsigned short* Apre = (unsigned short*)d_ws;
    unsigned short* Xhi  = (unsigned short*)((char*)d_ws + (1u << 20));
    unsigned short* Xlo  = (unsigned short*)((char*)d_ws + (2u << 20));

    build_g_kernel<<<256, 256, 0, stream>>>(W, Apre);
    split_x_kernel<<<2048, 256, 0, stream>>>(inp, Xhi, Xlo);

    dim3 grid(512, 16);
    kspace_gemm_kernel<<<grid, 256, 0, stream>>>(Apre, Xhi, Xlo, out);

    copy_others_kernel<<<512, 256, 0, stream>>>(inp, out);
}
